// GraphAttention_44933947850904
// MI455X (gfx1250) — matrix operand order repaired and checked by reading, not compiled, not run on hardware
//
#include <hip/hip_runtime.h>
#include <hip/hip_bf16.h>

typedef _Float16 half_t;
typedef __attribute__((ext_vector_type(16))) _Float16 v16h;
typedef __attribute__((ext_vector_type(8)))  _Float16 v8h;
typedef __attribute__((ext_vector_type(8)))  float    v8f;

#define GN  4096
#define GF  256
#define GFO 128
#define GH  4
#define BSTRIDE 40   // padded LDS row stride in halves (80B, multiple of 16B)

__device__ __forceinline__ float sigmoidf_fast(float x) {
    return 1.0f / (1.0f + __expf(-x));
}

struct h2pair { half_t x, y; };

// ---------------------------------------------------------------------------
// Kernel 1: feats[h] = X @ W[h] + b[h]   via v_wmma_f32_16x16x32_f16
// grid (GN/64, GH), block 128 (4 waves, each wave: 16 rows x 128 cols)
// ---------------------------------------------------------------------------
__global__ __launch_bounds__(128)
void k_feats(const float* __restrict__ X, const float* __restrict__ W,
             const float* __restrict__ b,
             float* __restrict__ feats_f32, half_t* __restrict__ feats_f16)
{
    const int h    = blockIdx.y;
    const int row0 = blockIdx.x * 64;
    const int t    = threadIdx.x;
    const int wave = t >> 5;
    const int lane = t & 31;
    const int lo16 = lane & 15;
    const int hi   = lane >> 4;

    __shared__ half_t lds_wT[GFO * BSTRIDE];   // W k-slice, transposed [n][k]

    v8f C[8] = {};
    const float* Wh   = W + (size_t)h * GF * GFO;
    const int    mrow = row0 + wave * 16 + lo16;

    for (int k0 = 0; k0 < GF; k0 += 32) {
        __syncthreads();
        // stage W[k0..k0+31][0..127] transposed into LDS as f16 (coalesced read)
        #pragma unroll
        for (int r = 0; r < 32; ++r) {
            int d = r * 128 + t;           // 0..4095
            int k = d >> 7;                // 0..31
            int n = d & 127;
            lds_wT[n * BSTRIDE + k] = (half_t)Wh[(size_t)(k0 + k) * GFO + n];
        }
        __syncthreads();

        // A operand: 16-bit A 16x32 layout (ISA 7.12.2)
        v16h a;
        const float* xr = X + (size_t)mrow * GF + k0;
        #pragma unroll
        for (int i = 0; i < 16; ++i) {
            int kk = (i < 8) ? (hi * 8 + i) : (8 + hi * 8 + i);  // K index
            a[i] = (half_t)xr[kk];
        }

        #pragma unroll
        for (int nt = 0; nt < 8; ++nt) {
            int n = nt * 16 + lo16;
            // B operand: lanes 0-15 hold K=0-7,16-23, lanes 16-31 hold K=8-15,24-31
            v8h blo = *(const v8h*)&lds_wT[n * BSTRIDE + hi * 8];
            v8h bhi = *(const v8h*)&lds_wT[n * BSTRIDE + 16 + hi * 8];
            v16h bb;
            #pragma unroll
            for (int i = 0; i < 8; ++i) { bb[i] = blo[i]; bb[i + 8] = bhi[i]; }
            C[nt] = __builtin_amdgcn_wmma_f32_16x16x32_f16(
                        false, a, false, bb, (short)0, C[nt], false, false);
        }
    }

    // epilogue: +bias, store f32 and f16 copies
    #pragma unroll
    for (int nt = 0; nt < 8; ++nt) {
        int   n    = nt * 16 + lo16;
        float bias = b[h * GFO + n];
        #pragma unroll
        for (int r = 0; r < 8; ++r) {
            int   row = row0 + wave * 16 + r + 8 * hi;   // C layout: M = r + 8*hi
            float v   = C[nt][r] + bias;
            size_t off = ((size_t)h * GN + row) * GFO + n;
            feats_f32[off] = v;
            feats_f16[off] = (half_t)v;
        }
    }
}

// ---------------------------------------------------------------------------
// Kernel 2: attention logits  a_s[h][i], a_n[h][i]  (one wave per head-row)
// grid GH*GN/4 blocks, block 128
// ---------------------------------------------------------------------------
__global__ __launch_bounds__(128)
void k_logits(const float* __restrict__ feats_f32,
              const float* __restrict__ ak_self, const float* __restrict__ ak_neigh,
              const float* __restrict__ ab_self, const float* __restrict__ ab_neigh,
              float* __restrict__ a_s, float* __restrict__ a_n)
{
    const int t    = threadIdx.x;
    const int wave = t >> 5;
    const int lane = t & 31;
    const int g    = blockIdx.x * 4 + wave;   // g = h*GN + row
    const int h    = g >> 12;

    const float4 f  = *(const float4*)&feats_f32[(size_t)g * GFO + lane * 4];
    const float4 s4 = *(const float4*)&ak_self [h * GFO + lane * 4];
    const float4 n4 = *(const float4*)&ak_neigh[h * GFO + lane * 4];
    float s  = f.x * s4.x + f.y * s4.y + f.z * s4.z + f.w * s4.w;
    float nn = f.x * n4.x + f.y * n4.y + f.z * n4.z + f.w * n4.w;
    #pragma unroll
    for (int off = 16; off >= 1; off >>= 1) {
        s  += __shfl_xor(s,  off, 32);
        nn += __shfl_xor(nn, off, 32);
    }
    if (lane == 0) {
        a_s[g] = s  + ab_self[h];
        a_n[g] = nn + ab_neigh[h];
    }
}

// ---------------------------------------------------------------------------
// Kernel 3: node_feats = sigmoid-gated att @ feats + feats, ReLU, concat-out
// grid (GN/64, GH), block 128 (4 waves; wave w owns rows [i0+16w, i0+16w+16),
// each wave accumulates all 8 N-tiles -> 32 WMMAs per K-step per block)
// ---------------------------------------------------------------------------
__global__ __launch_bounds__(128)
void k_aggregate(const float* __restrict__ A,
                 const float* __restrict__ a_s, const float* __restrict__ a_n,
                 const half_t* __restrict__ feats_f16,
                 const float* __restrict__ feats_f32,
                 float* __restrict__ out)
{
    const int h    = blockIdx.y;
    const int i0   = blockIdx.x * 64;
    const int t    = threadIdx.x;
    const int wave = t >> 5;
    const int lane = t & 31;
    const int lo16 = lane & 15;
    const int hi   = lane >> 4;

    __shared__ half_t lds_att[64 * BSTRIDE];    // att tile [m][j], f16
    __shared__ half_t lds_bT [GFO * BSTRIDE];   // feats slice, transposed [n][k]

    v8f C[8] = {};

    // att staging assignment: thread t -> row am = t>>1, 16 cols at ac
    const int    am   = t >> 1;          // 0..63
    const int    ac   = (t & 1) * 16;    // 0 or 16
    const float  as_m = a_s[h * GN + i0 + am];
    const float* an_h = a_n + h * GN;
    const half_t* fh  = feats_f16 + (size_t)h * GN * GFO;

    for (int j0 = 0; j0 < GN; j0 += 32) {
        __syncthreads();
        // ---- stage 64x32 attention tile (f16) ----
        {
            const float* arow = &A[(size_t)(i0 + am) * GN + j0 + ac];
            const float* anp  = &an_h[j0 + ac];
            half_t* dst = &lds_att[am * BSTRIDE + ac];
            #pragma unroll
            for (int c = 0; c < 16; c += 4) {
                float4 a4  = *(const float4*)&arow[c];
                float4 an4 = *(const float4*)&anp[c];
                dst[c + 0] = (half_t)((a4.x != 0.0f) ? sigmoidf_fast(as_m + an4.x) : 0.0f);
                dst[c + 1] = (half_t)((a4.y != 0.0f) ? sigmoidf_fast(as_m + an4.y) : 0.0f);
                dst[c + 2] = (half_t)((a4.z != 0.0f) ? sigmoidf_fast(as_m + an4.z) : 0.0f);
                dst[c + 3] = (half_t)((a4.w != 0.0f) ? sigmoidf_fast(as_m + an4.w) : 0.0f);
            }
            __builtin_prefetch(arow + 32, 0, 0);   // next adjacency tile
        }
        // ---- stage feats[j0..j0+31][0..127] transposed into LDS ----
        #pragma unroll
        for (int r = 0; r < 16; ++r) {
            int d  = r * 128 + t;        // 0..2047 dword slots
            int k  = d >> 6;             // 0..31
            int np = (d & 63) * 2;       // n pair base
            h2pair p = *(const h2pair*)&fh[(size_t)(j0 + k) * GFO + np];
            lds_bT[ np      * BSTRIDE + k] = p.x;
            lds_bT[(np + 1) * BSTRIDE + k] = p.y;
        }
        __syncthreads();

        // ---- A operand (this wave's 16 att rows), ISA 16-bit A 16x32 layout ----
        v16h a;
        {
            const int mrow = wave * 16 + lo16;
            v8h alo = *(const v8h*)&lds_att[mrow * BSTRIDE + hi * 8];
            v8h ahi = *(const v8h*)&lds_att[mrow * BSTRIDE + 16 + hi * 8];
            #pragma unroll
            for (int i = 0; i < 8; ++i) { a[i] = alo[i]; a[i + 8] = ahi[i]; }
        }
        // ---- all 8 N-tiles ----
        #pragma unroll
        for (int nt = 0; nt < 8; ++nt) {
            int n = nt * 16 + lo16;
            v8h blo = *(const v8h*)&lds_bT[n * BSTRIDE + hi * 8];
            v8h bhi = *(const v8h*)&lds_bT[n * BSTRIDE + 16 + hi * 8];
            v16h bb;
            #pragma unroll
            for (int i = 0; i < 8; ++i) { bb[i] = blo[i]; bb[i + 8] = bhi[i]; }
            C[nt] = __builtin_amdgcn_wmma_f32_16x16x32_f16(
                        false, a, false, bb, (short)0, C[nt], false, false);
        }
    }

    // ---- epilogue: residual + ReLU, scatter into concatenated out ----
    const float* ff = feats_f32 + (size_t)h * GN * GFO;
    #pragma unroll
    for (int nt = 0; nt < 8; ++nt) {
        int n = nt * 16 + lo16;
        #pragma unroll
        for (int r = 0; r < 8; ++r) {
            int   row = i0 + wave * 16 + r + 8 * hi;
            float v   = C[nt][r] + ff[(size_t)row * GFO + n];
            out[(size_t)row * (GH * GFO) + h * GFO + n] = fmaxf(v, 0.0f);
        }
    }
}

// ---------------------------------------------------------------------------
// Kernel 4: attn = mean over heads of sigmoid-gated attention (streaming)
// ---------------------------------------------------------------------------
__global__ __launch_bounds__(256)
void k_attn_mean(const float* __restrict__ A,
                 const float* __restrict__ a_s, const float* __restrict__ a_n,
                 float* __restrict__ attn)
{
    const size_t idx = ((size_t)blockIdx.x * 256 + threadIdx.x) * 4;
    const int i = (int)(idx >> 12);
    const int j = (int)(idx & 4095);

    float4 a4 = *(const float4*)&A[idx];
    float  asi[GH];
    float4 anj[GH];
    #pragma unroll
    for (int h = 0; h < GH; ++h) {
        asi[h] = a_s[h * GN + i];
        anj[h] = *(const float4*)&a_n[h * GN + j];
    }
    float4 r;
    r.x = 0.0f; r.y = 0.0f; r.z = 0.0f; r.w = 0.0f;
    if (a4.x != 0.0f) {
        #pragma unroll
        for (int h = 0; h < GH; ++h) r.x += sigmoidf_fast(asi[h] + anj[h].x);
    }
    if (a4.y != 0.0f) {
        #pragma unroll
        for (int h = 0; h < GH; ++h) r.y += sigmoidf_fast(asi[h] + anj[h].y);
    }
    if (a4.z != 0.0f) {
        #pragma unroll
        for (int h = 0; h < GH; ++h) r.z += sigmoidf_fast(asi[h] + anj[h].z);
    }
    if (a4.w != 0.0f) {
        #pragma unroll
        for (int h = 0; h < GH; ++h) r.w += sigmoidf_fast(asi[h] + anj[h].w);
    }
    r.x *= 0.25f; r.y *= 0.25f; r.z *= 0.25f; r.w *= 0.25f;
    *(float4*)&attn[idx] = r;
}

// ---------------------------------------------------------------------------
extern "C" void kernel_launch(void* const* d_in, const int* in_sizes, int n_in,
                              void* d_out, int out_size, void* d_ws, size_t ws_size,
                              hipStream_t stream) {
    (void)in_sizes; (void)n_in; (void)out_size; (void)ws_size;

    const float* X        = (const float*)d_in[0];
    const float* A        = (const float*)d_in[1];
    const float* W        = (const float*)d_in[2];
    const float* b        = (const float*)d_in[3];
    const float* ak_self  = (const float*)d_in[4];
    const float* ak_neigh = (const float*)d_in[5];
    const float* ab_self  = (const float*)d_in[6];
    const float* ab_neigh = (const float*)d_in[7];

    float* out  = (float*)d_out;                              // [4096, 512]
    float* attn = (float*)d_out + (size_t)GN * GH * GFO;      // [4096, 4096]

    char*   ws        = (char*)d_ws;
    float*  feats_f32 = (float*)ws;                                        // 8 MB
    half_t* feats_f16 = (half_t*)(ws + (size_t)GH * GN * GFO * 4);         // 4 MB
    float*  a_s       = (float*)(ws + (size_t)GH * GN * GFO * 6);          // 64 KB
    float*  a_n       = a_s + GH * GN;                                     // 64 KB

    k_feats<<<dim3(GN / 64, GH), 128, 0, stream>>>(X, W, b, feats_f32, feats_f16);
    k_logits<<<(GH * GN) / 4, 128, 0, stream>>>(feats_f32, ak_self, ak_neigh,
                                                ab_self, ab_neigh, a_s, a_n);
    k_aggregate<<<dim3(GN / 64, GH), 128, 0, stream>>>(A, a_s, a_n,
                                                       feats_f16, feats_f32, out);
    k_attn_mean<<<(GN / 64) * (GN / 256), 256, 0, stream>>>(A, a_s, a_n, attn);
}
